// HyperLoRADense_80092550136286
// MI455X (gfx1250) — matrix operand order repaired and checked by reading, not compiled, not run on hardware
//
#include <hip/hip_runtime.h>

#define B_  4
#define S_  4096
#define D_  1024
#define F_  1024
#define R_  4
#define H_  512
#define KC_ (D_ / 32)   // 32 k-chunks
#define NT_ (F_ / 16)   // 64 n-tiles

typedef __attribute__((ext_vector_type(16))) __bf16        v16bf;
typedef __attribute__((ext_vector_type(8)))  float         v8f;
typedef __attribute__((ext_vector_type(4)))  unsigned int  u32x4;

union FragBF {
  v16bf v;
  u32x4 q[2];
  unsigned short u[16];
};

__device__ __forceinline__ unsigned short f32_to_bf16_rne(float f) {
  unsigned int u = __float_as_uint(f);
  u += 0x7FFFu + ((u >> 16) & 1u);
  return (unsigned short)(u >> 16);
}
__device__ __forceinline__ float bf16_bits_to_f32(unsigned short h) {
  return __uint_as_float(((unsigned int)h) << 16);
}

// ---------------------------------------------------------------------------
// Kernel 1: lora_a = (hyper @ Wa) * 1/64, lora_b = (hyper @ Wb) * 1/64
// ---------------------------------------------------------------------------
__global__ void lora_factors_kernel(const float* __restrict__ hv,
                                    const float* __restrict__ Wa,
                                    const float* __restrict__ Wb,
                                    float* __restrict__ la,
                                    float* __restrict__ lb) {
  const int t  = blockIdx.x * blockDim.x + threadIdx.x;   // 0 .. 32767
  const int NA = B_ * R_ * D_;                            // 16384
  const bool isA = (t < NA);
  const int tt   = isA ? t : t - NA;
  const int cols = R_ * D_;                               // 4096 (== R_*F_)
  const int b    = tt / cols;
  const int col  = tt % cols;
  const float* Wsel = isA ? Wa : Wb;
  const float* hrow = hv + b * H_;
  float acc = 0.f;
  for (int h = 0; h < H_; ++h)
    acc += hrow[h] * Wsel[(size_t)h * cols + col];
  (isA ? la : lb)[tt] = acc * 0.015625f;   // sqrt(1/(1024*4)) == 1/64
}

// ---------------------------------------------------------------------------
// Kernel 2: W_eff = W + la^T @ lb, split to bf16 hi/lo, stored PRE-SWIZZLED
// in WMMA B-fragment lane order:
//   element e of lane L of tile (b, ntile, kc) holds
//     W_eff[b][ d = kc*32 + 16*(L>>4) + e ][ f = ntile*16 + (L&15) ]
//   at ushort index ((((b*NT + ntile)*KC + kc)*32 + L)*16 + e
// so the GEMM loads each fragment as two global b128 reads.
// Block handles d-range 32 (one kc) x f-range 128 (8 ntiles) via LDS transpose.
// ---------------------------------------------------------------------------
__global__ __launch_bounds__(256) void weff_swizzle_kernel(
    const float* __restrict__ W,
    const float* __restrict__ la,
    const float* __restrict__ lb,
    unsigned short* __restrict__ whi,
    unsigned short* __restrict__ wlo) {
  __shared__ __align__(16) unsigned short sHi[32][128];
  __shared__ __align__(16) unsigned short sLo[32][128];

  const int b  = blockIdx.z;
  const int kc = blockIdx.y;            // 0..31
  const int fg = blockIdx.x;            // 0..7
  const int d0 = kc * 32;
  const int f0 = fg * 128;
  const int tid = threadIdx.x;

  // Phase 1: coalesced read of W rows, add rank-4 update, split, stage to LDS
#pragma unroll
  for (int it = 0; it < 4; ++it) {
    const int q  = tid + it * 256;      // 0..1023 (32 rows x 32 float4)
    const int dd = q >> 5;              // 0..31
    const int cc = (q & 31) << 2;       // 0..124
    const int d  = d0 + dd;
    const float4 w4 = *(const float4*)(W + (size_t)d * F_ + f0 + cc);
    float lar[R_];
#pragma unroll
    for (int r = 0; r < R_; ++r) lar[r] = la[((b * R_) + r) * D_ + d];
    const float wv[4] = {w4.x, w4.y, w4.z, w4.w};
#pragma unroll
    for (int e = 0; e < 4; ++e) {
      const int f = f0 + cc + e;
      float v = wv[e];
#pragma unroll
      for (int r = 0; r < R_; ++r) v += lar[r] * lb[((b * R_) + r) * F_ + f];
      const unsigned short hi = f32_to_bf16_rne(v);
      sHi[dd][cc + e] = hi;
      sLo[dd][cc + e] = f32_to_bf16_rne(v - bf16_bits_to_f32(hi));
    }
  }
  __syncthreads();

  // Phase 2: gather lane fragments from LDS, write coalesced b128 runs
  const int ntl   = tid >> 5;           // 0..7 local n-tile
  const int lane  = tid & 31;
  const int laneM = lane & 15;
  const int halfL = lane >> 4;
  FragBF hi, lo;
#pragma unroll
  for (int e = 0; e < 16; ++e) {
    const int dd = (halfL << 4) + e;
    const int ff = ntl * 16 + laneM;
    hi.u[e] = sHi[dd][ff];
    lo.u[e] = sLo[dd][ff];
  }
  const size_t ntile = (size_t)fg * 8 + ntl;
  const size_t base  = ((((size_t)b * NT_ + ntile) * KC_ + kc) * 32 + lane) * 16;
  *(u32x4*)(whi + base)     = hi.q[0];
  *(u32x4*)(whi + base + 8) = hi.q[1];
  *(u32x4*)(wlo + base)     = lo.q[0];
  *(u32x4*)(wlo + base + 8) = lo.q[1];
}

// ---------------------------------------------------------------------------
// Kernel 3: batched GEMM  out[b] = x[b] @ W_eff[b] + bias
// 128x128 C tile per block (8 waves, each 32x64). A staged fp32->bf16 hi/lo
// in LDS; B fragments loaded directly from pre-swizzled global (L2-resident).
// Split-bf16: acc += Ahi*Bhi + Ahi*Blo + Alo*Bhi (fp32 accumulate).
// ---------------------------------------------------------------------------
__global__ __launch_bounds__(256) void hyperlora_gemm_kernel(
    const float* __restrict__ x, const float* __restrict__ bias,
    const unsigned short* __restrict__ whi, const unsigned short* __restrict__ wlo,
    float* __restrict__ out) {
  __shared__ __align__(16) unsigned short sAhi[128][34];   // [m][k], pad 2
  __shared__ __align__(16) unsigned short sAlo[128][34];

  const int b    = blockIdx.z;
  const int m0   = blockIdx.x * 128;
  const int tid  = threadIdx.x;
  const int lane = tid & 31;
  const int wave = tid >> 5;
  const int wm   = wave & 3;       // wave's 32-row band
  const int wn   = wave >> 2;      // wave's 64-col band
  const int laneM = lane & 15;
  const int halfL = lane >> 4;

  const float* xT = x + (size_t)(b * S_ + m0) * D_;
  // first swizzled n-tile for this wave; fragment stride between kc's:
  const int ntb = blockIdx.y * 8 + wn * 4;
  const unsigned short* bhiT[4];
  const unsigned short* bloT[4];
#pragma unroll
  for (int nt = 0; nt < 4; ++nt) {
    const size_t off = ((((size_t)b * NT_ + (ntb + nt)) * KC_) * 32 + lane) * 16;
    bhiT[nt] = whi + off;
    bloT[nt] = wlo + off;
  }

  v8f acc[2][4] = {};

  for (int kc = 0; kc < KC_; ++kc) {
    const int k0 = kc * 32;
    // ---- stage A: fp32 x -> split bf16 hi/lo in LDS (128x32) ----
#pragma unroll
    for (int it = 0; it < 4; ++it) {
      const int q   = tid + it * 256;        // 0..1023
      const int row = q >> 3;
      const int c   = (q & 7) << 2;
      const float4 v = *(const float4*)(xT + (size_t)row * D_ + k0 + c);
      const float vv[4] = {v.x, v.y, v.z, v.w};
#pragma unroll
      for (int e = 0; e < 4; ++e) {
        const unsigned short hi = f32_to_bf16_rne(vv[e]);
        sAhi[row][c + e] = hi;
        sAlo[row][c + e] = f32_to_bf16_rne(vv[e] - bf16_bits_to_f32(hi));
      }
    }
    if (kc + 1 < KC_) {   // uniform: prefetch next kc's B fragments toward L2
      __builtin_prefetch(bhiT[0] + (size_t)(kc + 1) * 512, 0, 1);
      __builtin_prefetch(bloT[0] + (size_t)(kc + 1) * 512, 0, 1);
    }
    __syncthreads();

    // ---- A fragments from LDS (contiguous 8-ushort runs -> merged ds) ----
    FragBF ahi[2], alo[2];
#pragma unroll
    for (int mt = 0; mt < 2; ++mt) {
      const int m = wm * 32 + mt * 16 + laneM;
#pragma unroll
      for (int e = 0; e < 16; ++e) {
        const int kk = (halfL << 4) + e;
        ahi[mt].u[e] = sAhi[m][kk];
        alo[mt].u[e] = sAlo[m][kk];
      }
    }

    // ---- B fragments: 2x b128 direct from swizzled global; 3 WMMAs each ----
#pragma unroll
    for (int nt = 0; nt < 4; ++nt) {
      FragBF bhi, blo;
      const unsigned short* ph = bhiT[nt] + (size_t)kc * 512;  // 32 lanes*16
      const unsigned short* pl = bloT[nt] + (size_t)kc * 512;
      bhi.q[0] = *(const u32x4*)(ph);
      bhi.q[1] = *(const u32x4*)(ph + 8);
      blo.q[0] = *(const u32x4*)(pl);
      blo.q[1] = *(const u32x4*)(pl + 8);
#pragma unroll
      for (int mt = 0; mt < 2; ++mt) {
        acc[mt][nt] = __builtin_amdgcn_wmma_f32_16x16x32_bf16(
            false, ahi[mt].v, false, bhi.v, (short)0, acc[mt][nt], false, false);
        acc[mt][nt] = __builtin_amdgcn_wmma_f32_16x16x32_bf16(
            false, ahi[mt].v, false, blo.v, (short)0, acc[mt][nt], false, false);
        acc[mt][nt] = __builtin_amdgcn_wmma_f32_16x16x32_bf16(
            false, alo[mt].v, false, bhi.v, (short)0, acc[mt][nt], false, false);
      }
    }
    __syncthreads();
  }

  // ---- epilogue: C/D layout VGPR v -> M = 8*half + v, N = lane%16 ----
  const int n0 = blockIdx.y * 128;
#pragma unroll
  for (int mt = 0; mt < 2; ++mt)
#pragma unroll
    for (int nt = 0; nt < 4; ++nt) {
      const int n  = n0 + wn * 64 + nt * 16 + laneM;
      const float bv = bias[n];
#pragma unroll
      for (int v = 0; v < 8; ++v) {
        const int m = m0 + wm * 32 + mt * 16 + (halfL << 3) + v;
        out[(size_t)(b * S_ + m) * F_ + n] = acc[mt][nt][v] + bv;
      }
    }
}

// ---------------------------------------------------------------------------
extern "C" void kernel_launch(void* const* d_in, const int* in_sizes, int n_in,
                              void* d_out, int out_size, void* d_ws, size_t ws_size,
                              hipStream_t stream) {
  const float* x    = (const float*)d_in[0];
  const float* hv   = (const float*)d_in[1];
  const float* W    = (const float*)d_in[2];
  const float* bias = (const float*)d_in[3];
  const float* Wa   = (const float*)d_in[4];
  const float* Wb   = (const float*)d_in[5];
  float* out = (float*)d_out;

  // workspace: la (64KB) | lb (64KB) | whi (8MB, swizzled) | wlo (8MB, swizzled)
  float* la = (float*)d_ws;
  float* lb = la + B_ * R_ * D_;
  unsigned short* whi = (unsigned short*)(lb + B_ * R_ * F_);
  unsigned short* wlo = whi + (size_t)B_ * D_ * F_;

  lora_factors_kernel<<<(B_ * R_ * (D_ + F_)) / 256, 256, 0, stream>>>(hv, Wa, Wb, la, lb);

  dim3 gSwz(F_ / 128, KC_, B_);
  weff_swizzle_kernel<<<gSwz, 256, 0, stream>>>(W, la, lb, whi, wlo);

  dim3 gGemm(S_ / 128, F_ / 128, B_);
  hyperlora_gemm_kernel<<<gGemm, 256, 0, stream>>>(x, bias, whi, wlo, out);
}
